// OrientedCoordinateAttention_88321707475342
// MI455X (gfx1250) — matrix operand order repaired and checked by reading, not compiled, not run on hardware
//
#include <hip/hip_runtime.h>
#include <hip/hip_bf16.h>

typedef __attribute__((ext_vector_type(16))) _Float16 v16h;
typedef __attribute__((ext_vector_type(8)))  _Float16 v8h;
typedef __attribute__((ext_vector_type(8)))  float    v8f;

#define Bn 8
#define Cn 64
#define Hn 256
#define Wn 256
#define HID 8
#define KC (Cn * 9)          // 576
#define PI_F 3.14159265358979f

__device__ __forceinline__ float sigmoidf_(float v) {
    return 1.0f / (1.0f + __expf(-v));
}

// Build a v16h fragment from two 16-byte LDS loads (8 halves each).
__device__ __forceinline__ v16h ld_frag(const _Float16* lo, const _Float16* hi) {
    v8h l = *(const v8h*)lo;
    v8h h = *(const v8h*)hi;
    return __builtin_shufflevector(l, h, 0,1,2,3,4,5,6,7,8,9,10,11,12,13,14,15);
}

// ---------------------------------------------------------------------------
// Kernel 1: theta = pi*sigmoid(conv1x1(relu(conv3x3(x)))) ; store cos/sin.
// Block = 128 threads = 4 waves, covers (b, y, 64-pixel x tile).
// Input patch (3 rows x 66 cols x 64 ch) staged in LDS as f16 in
// [spatial][channel] order; K reordered tap-major (k = tap*64 + c) so every
// WMMA fragment is two contiguous ds_load_b128 per lane. 18 WMMA per wave.
// ---------------------------------------------------------------------------
__global__ __launch_bounds__(128) void k_theta(
    const float* __restrict__ x,  const float* __restrict__ w3,
    const float* __restrict__ b3, const float* __restrict__ w1,
    const float* __restrict__ b1, float* __restrict__ cosT,
    float* __restrict__ sinT)
{
    __shared__ alignas(16) _Float16 patch[3 * 66 * Cn];   // 25344 B
    __shared__ alignas(16) _Float16 w3r[16 * KC];         // 18432 B, rows 8..15 = 0
    __shared__ float hidLds[4][16][16];                   // 4096 B

    const int t   = threadIdx.x;
    const int bid = blockIdx.x;
    const int tilex = bid & 3;               // W/64 = 4 tiles per row
    const int y     = (bid >> 2) & (Hn - 1);
    const int b     = bid >> 10;
    const int x0    = tilex * 64;
    const long planeB = (long)b * Cn * Hn * Wn;

    // ---- fill reordered w3 (tap-major K), zero-padded to 16 output rows ---
    for (int i = t; i < 16 * KC; i += 128) {
        int n = i / KC;
        int k = i - n * KC;
        _Float16 v = (_Float16)0.0f;
        if (n < HID) {
            int j = k >> 6;          // tap 0..8
            int c = k & 63;          // channel
            v = (_Float16)w3[n * KC + c * 9 + j];
        }
        w3r[i] = v;
    }

    // ---- fill input patch: coalesced along x, borders zeroed --------------
    {
        const int g  = t >> 6;       // 0..1 (channel parity)
        const int l64 = t & 63;      // column lane
        for (int r = 0; r < 3; ++r) {
            int yy = y + r - 1;
            bool yok = (unsigned)yy < (unsigned)Hn;
            for (int c = g; c < Cn; c += 2) {
                const float* src = x + planeB + ((long)c * Hn + yy) * Wn;
                {
                    int col = l64;
                    int xx = x0 - 1 + col;
                    float v = (yok && (unsigned)xx < (unsigned)Wn) ? src[xx] : 0.0f;
                    patch[(r * 66 + col) * Cn + c] = (_Float16)v;
                }
                if (l64 < 2) {
                    int col = 64 + l64;
                    int xx = x0 - 1 + col;
                    float v = (yok && (unsigned)xx < (unsigned)Wn) ? src[xx] : 0.0f;
                    patch[(r * 66 + col) * Cn + c] = (_Float16)v;
                }
            }
        }
    }
    __syncthreads();

    const int wave = t >> 5;
    const int lane = t & 31;
    const int m    = lane & 15;                 // A row (pixel)
    const int kbA  = (lane < 16) ? 0 : 8;       // A K-interleave base
    const int nB   = lane & 15;                 // B column
    const int kbB  = (lane < 16) ? 0 : 8;       // B K base

    v8f acc = {};
    #pragma unroll
    for (int q = 0; q < 18; ++q) {
        const int j  = q >> 1;                  // tap (constant after unroll)
        const int ky = j / 3;
        const int kx = j - ky * 3;
        const int cb = (q & 1) * 32 + kbA;      // channel base within chunk
        const _Float16* ap = &patch[(ky * 66 + wave * 16 + m + kx) * Cn + cb];
        v16h a = ld_frag(ap, ap + 16);
        const _Float16* bp = &w3r[nB * KC + q * 32 + kbB];
        v16h bm = ld_frag(bp, bp + 16);
        acc = __builtin_amdgcn_wmma_f32_16x16x32_f16(
                  false, a, false, bm, (short)0, acc, false, false);
    }

    // D layout: lane holds N = lane&15; VGPR r -> M = r + (lane<16 ? 0 : 8)
    {
        int n    = lane & 15;
        int mofs = (lane < 16) ? 0 : 8;
        #pragma unroll
        for (int r = 0; r < 8; ++r)
            hidLds[wave][r + mofs][n] = acc[r];
    }
    __syncthreads();

    if (lane < 16) {
        int p = lane;
        float s = b1[0];
        #pragma unroll
        for (int n = 0; n < HID; ++n) {
            float h = hidLds[wave][p][n] + b3[n];
            s += fmaxf(h, 0.0f) * w1[n];
        }
        float theta = PI_F * sigmoidf_(s);
        int idx = (b * Hn + y) * Wn + (x0 + wave * 16 + p);
        cosT[idx] = __cosf(theta);
        sinT[idx] = __sinf(theta);
    }
}

// ---------------------------------------------------------------------------
// Kernel 2: fused oriented pooling + per-pixel MLP (WMMA) + output scale.
// Block = 128 threads per (b, y, 16-pixel tile). Pooling: thread (p=t&15,
// g=t>>4) handles 16 channels of one direction -> coalesced gathers.
// MLP: all 4 waves run layer 1 (K=128, 4 WMMA); layer-2 N-tiles split
// 2 per wave. All fragment loads are unpredicated b128 from zero-padded LDS.
// ---------------------------------------------------------------------------
__global__ __launch_bounds__(128) void k_pool_mlp(
    const float* __restrict__ x,    const float* __restrict__ wr,
    const float* __restrict__ br,   const float* __restrict__ we,
    const float* __restrict__ be,   const float* __restrict__ cosT,
    const float* __restrict__ sinT, float* __restrict__ out)
{
    __shared__ alignas(16) _Float16 ctxH[16 * 128];   // context [pixel][2C]
    __shared__ alignas(16) _Float16 wrP[16 * 128];    // wr, rows 8..15 = 0
    __shared__ alignas(16) _Float16 weP[128 * 32];    // we, K padded 8->32
    __shared__ alignas(16) _Float16 hidP[16 * 32];    // relu(hidden), K pad 0
    __shared__ float    wbuf[16][128];
    __shared__ float    brS[HID], beS[128], cosS[16], sinS[16];

    const int t   = threadIdx.x;
    const int bid = blockIdx.x;
    const int tx  = bid & 15;
    const int y   = (bid >> 4) & (Hn - 1);
    const int b   = bid >> 12;
    const int x0  = tx * 16;

    for (int i = t; i < 16 * 128; i += 128) {
        int n = i >> 7, k = i & 127;
        wrP[i] = (n < HID) ? (_Float16)wr[n * 128 + k] : (_Float16)0.0f;
    }
    for (int i = t; i < 128 * 32; i += 128) {
        int o = i >> 5, k = i & 31;
        weP[i] = (k < HID) ? (_Float16)we[o * HID + k] : (_Float16)0.0f;
    }
    for (int i = t; i < 16 * 32; i += 128) hidP[i] = (_Float16)0.0f;
    beS[t] = be[t];
    if (t < HID) brS[t] = br[t];
    if (t < 16) {
        int idx = (b * Hn + y) * Wn + x0 + t;
        cosS[t] = cosT[idx];
        sinS[t] = sinT[idx];
    }
    __syncthreads();

    // ---- oriented pooling -> ctxH -----------------------------------------
    {
        const int p     = t & 15;
        const int g     = t >> 4;       // 0..7
        const int dir   = g >> 2;       // 0 = tangential, 1 = normal
        const int cbase = (g & 3) * 16;
        const float ct = cosS[p], st = sinS[p];
        const float vx = dir ? -st : ct;
        const float vy = dir ?  ct : st;
        const float fy = (float)y;
        const float fx = (float)(x0 + p);

        int   o00[9], o01[9], o10[9], o11[9];
        float w00[9], w01[9], w10[9], w11[9];
        #pragma unroll
        for (int j = 0; j < 9; ++j) {
            float tt = (float)(j - 4);
            float cy = fminf(fmaxf(fy + tt * vy, 0.0f), 255.0f);
            float cx = fminf(fmaxf(fx + tt * vx, 0.0f), 255.0f);
            int y0i = (int)cy;
            int x0i = (int)cx;
            int y1i = min(y0i + 1, Hn - 1);
            int x1i = min(x0i + 1, Wn - 1);
            float wy = cy - (float)y0i;
            float wx = cx - (float)x0i;
            o00[j] = y0i * Wn + x0i;  o01[j] = y0i * Wn + x1i;
            o10[j] = y1i * Wn + x0i;  o11[j] = y1i * Wn + x1i;
            w00[j] = (1.0f - wy) * (1.0f - wx);  w01[j] = (1.0f - wy) * wx;
            w10[j] = wy * (1.0f - wx);           w11[j] = wy * wx;
        }
        #pragma unroll 2
        for (int i = 0; i < 16; ++i) {
            int c = cbase + i;
            const float* plane = x + (long)(b * Cn + c) * (Hn * Wn);
            float acc = 0.0f;
            #pragma unroll
            for (int j = 0; j < 9; ++j) {
                acc += plane[o00[j]] * w00[j] + plane[o01[j]] * w01[j]
                     + plane[o10[j]] * w10[j] + plane[o11[j]] * w11[j];
            }
            acc *= (1.0f / 9.0f);
            ctxH[p * 128 + dir * 64 + c] = (_Float16)acc;
        }
    }
    __syncthreads();

    const int wave = t >> 5;
    const int lane = t & 31;
    const int m    = lane & 15;
    const int kbA  = (lane < 16) ? 0 : 8;
    const int nB   = lane & 15;
    const int kbB  = (lane < 16) ? 0 : 8;

    // ---- MLP layer 1: [16x128] x [128x8] (all waves, K in 4 chunks) -------
    {
        v8f acc = {};
        #pragma unroll
        for (int q = 0; q < 4; ++q) {
            const _Float16* ap = &ctxH[m * 128 + q * 32 + kbA];
            v16h a = ld_frag(ap, ap + 16);
            const _Float16* bp = &wrP[nB * 128 + q * 32 + kbB];
            v16h bm = ld_frag(bp, bp + 16);
            acc = __builtin_amdgcn_wmma_f32_16x16x32_f16(
                      false, a, false, bm, (short)0, acc, false, false);
        }
        int mofs = (lane < 16) ? 0 : 8;
        if (nB < HID) {                 // same values from all 4 waves: benign
            #pragma unroll
            for (int r = 0; r < 8; ++r) {
                float h = acc[r] + brS[nB];
                hidP[(r + mofs) * 32 + nB] = (_Float16)fmaxf(h, 0.0f);
            }
        }
    }
    __syncthreads();

    // ---- MLP layer 2: [16x32(K pad)] x [32x128], 2 N-tiles per wave -------
    {
        const _Float16* ap = &hidP[m * 32 + kbA];
        v16h a = ld_frag(ap, ap + 16);
        int mofs = (lane < 16) ? 0 : 8;
        #pragma unroll
        for (int jj = 0; jj < 2; ++jj) {
            int j = wave * 2 + jj;              // N tile 0..7
            const _Float16* bp = &weP[(j * 16 + nB) * 32 + kbB];
            v16h bm = ld_frag(bp, bp + 16);
            v8f d = {};
            d = __builtin_amdgcn_wmma_f32_16x16x32_f16(
                    false, a, false, bm, (short)0, d, false, false);
            int o = j * 16 + nB;
            #pragma unroll
            for (int r = 0; r < 8; ++r)
                wbuf[r + mofs][o] = sigmoidf_(d[r] + beS[o]);
        }
    }
    __syncthreads();

    // ---- output: (alpha_tan + alpha_norm) * x, coalesced -------------------
    {
        const int p = t & 15;
        const int g = t >> 4;
        #pragma unroll
        for (int jj = 0; jj < 8; ++jj) {
            int c = g * 8 + jj;              // 0..63
            long idx = ((long)(b * Cn + c) * Hn + y) * Wn + (x0 + p);
            out[idx] = (wbuf[p][c] + wbuf[p][64 + c]) * x[idx];
        }
    }
}

// ---------------------------------------------------------------------------
extern "C" void kernel_launch(void* const* d_in, const int* in_sizes, int n_in,
                              void* d_out, int out_size, void* d_ws, size_t ws_size,
                              hipStream_t stream) {
    const float* x  = (const float*)d_in[0];
    const float* w3 = (const float*)d_in[1];
    const float* b3 = (const float*)d_in[2];
    const float* w1 = (const float*)d_in[3];
    const float* b1 = (const float*)d_in[4];
    const float* wr = (const float*)d_in[5];
    const float* br = (const float*)d_in[6];
    const float* we = (const float*)d_in[7];
    const float* be = (const float*)d_in[8];
    float* out  = (float*)d_out;
    float* cosT = (float*)d_ws;                       // B*H*W f32
    float* sinT = cosT + (size_t)Bn * Hn * Wn;        // B*H*W f32

    k_theta<<<Bn * Hn * (Wn / 64), 128, 0, stream>>>(
        x, w3, b3, w1, b1, cosT, sinT);
    k_pool_mlp<<<Bn * Hn * (Wn / 16), 128, 0, stream>>>(
        x, wr, br, we, be, cosT, sinT, out);
}
